// PairModel_simprot_encoder_clf_AE_54649163874857
// MI455X (gfx1250) — matrix operand order repaired and checked by reading, not compiled, not run on hardware
//
#include <hip/hip_runtime.h>

// ---------------------------------------------------------------------------
// Problem constants (match reference)
// ---------------------------------------------------------------------------
#define BB    1024
#define DD    512
#define NN    50000
#define KTOP  32
#define INV_SCALE 0.044194173824159216f   // 1/sqrt(512)

typedef __bf16 bf16_t;
typedef __attribute__((ext_vector_type(16))) __bf16 bf16x16;
typedef __attribute__((ext_vector_type(8)))  __bf16 bf16x8;
typedef __attribute__((ext_vector_type(8)))  float  f32x8;

// ---------------------------------------------------------------------------
// bf16 <-> f32 helpers (round-to-nearest-even)
// ---------------------------------------------------------------------------
__device__ __forceinline__ unsigned short f2bf(float f) {
    union { float f; unsigned int u; } c; c.f = f;
    unsigned int u = c.u;
    unsigned int r = (u + 0x7FFFu + ((u >> 16) & 1u)) >> 16;
    return (unsigned short)r;
}
__device__ __forceinline__ float bf2f(unsigned short h) {
    union { unsigned int u; float f; } c; c.u = ((unsigned int)h) << 16;
    return c.f;
}

// ---------------------------------------------------------------------------
// f32 -> bf16 conversion (elementwise)
// ---------------------------------------------------------------------------
__global__ __launch_bounds__(256)
void k_f32_to_bf16(const float* __restrict__ in, unsigned short* __restrict__ out, size_t n) {
    size_t i = (size_t)blockIdx.x * blockDim.x + threadIdx.x;
    if (i < n) out[i] = f2bf(in[i]);
}

// ---------------------------------------------------------------------------
// concat [a_row | b_row] -> bf16, rows of 2*D
// ---------------------------------------------------------------------------
__global__ __launch_bounds__(256)
void k_concat_bf16(const float* __restrict__ a, const float* __restrict__ b,
                   unsigned short* __restrict__ out, int D) {
    int row = blockIdx.x;
    for (int d = threadIdx.x; d < D; d += blockDim.x) {
        out[(size_t)row * 2 * D + d]     = f2bf(a[(size_t)row * D + d]);
        out[(size_t)row * 2 * D + D + d] = f2bf(b[(size_t)row * D + d]);
    }
}

// ---------------------------------------------------------------------------
// Async copy of one A chunk (16 rows x 32 bf16 = 1KB) into LDS using the
// CDNA5 async-to-LDS engine. Threads 0..63 (waves 0,1; wave-uniform guard)
// each move one 16-byte segment: global_load_async_to_lds_b128.
// ---------------------------------------------------------------------------
__device__ __forceinline__ void async_copy_A(const bf16_t* __restrict__ X, int m0, int K,
                                             int k0, unsigned short* dstLDS, int tid) {
    if (tid < 64) {
        int row = tid >> 2;
        int seg = tid & 3;
        const char* src = (const char*)(X + (size_t)(m0 + row) * K + k0) + seg * 16;
        unsigned dst = (unsigned)(unsigned long long)(dstLDS + row * 32 + seg * 8);
        asm volatile("global_load_async_to_lds_b128 %0, %1, off"
                     :: "v"(dst), "v"(src) : "memory");
    }
}
__device__ __forceinline__ void wait_async_le(int n, int tid) {
    if (tid < 64) {
        if (n == 0) asm volatile("s_wait_asynccnt 0x0" ::: "memory");
        else        asm volatile("s_wait_asynccnt 0x1" ::: "memory");
    }
}

// ---------------------------------------------------------------------------
// WMMA GEMM:  Y[M,Nc] = X[M,K] @ W[Nc,K]^T (+bias) (+relu)
// bf16 in, fp32 accumulate via v_wmma_f32_16x16x32_bf16.
// Block = 256 threads = 8 waves; block tile 16(M) x 256(N); each wave owns
// a 16x32 slice (two accumulators -> two back-to-back WMMAs per A fragment).
// A tile staged once per block in double-buffered LDS via async-to-LDS;
// B fragments register double-buffered so loads overlap WMMA.
// ---------------------------------------------------------------------------
__global__ __launch_bounds__(256)
void k_wmma_gemm_bf16(const bf16_t* __restrict__ X, const bf16_t* __restrict__ W,
                      const float* __restrict__ bias,
                      float* __restrict__ Yf, unsigned short* __restrict__ Ybf,
                      int M, int Nc, int K, int relu) {
    __shared__ unsigned short Ash[2][16 * 32];   // 2KB double-buffered A chunk

    const int tid  = threadIdx.x;
    const int lane = tid & 31;
    const int wave = tid >> 5;
    const int m0 = blockIdx.y * 16;
    const int n0 = blockIdx.x * 256 + wave * 32;

    const int valid0 = (n0 + 16) <= Nc;
    const int valid1 = (n0 + 32) <= Nc;
    const int nt0 = valid0 ? n0      : 0;   // clamped bases keep loads in range
    const int nt1 = valid1 ? n0 + 16 : 0;

    const int r    = lane & 15;        // row within A tile / col within B,C tiles
    const int hsel = lane >> 4;        // lane half select
    const int ah   = hsel * 16;        // A K-offset (lanes16-31 hold K=16..31)
    const int bh   = hsel * 16;        // B K-offset (lanes16-31 hold K=16..31)

    const bf16_t* __restrict__ brow0 = W + (size_t)(nt0 + r) * K;
    const bf16_t* __restrict__ brow1 = W + (size_t)(nt1 + r) * K;

    // prologue: kick off A chunk 0, preload B fragments for k0 = 0
    async_copy_A(X, m0, K, 0, Ash[0], tid);
    bf16x8 cb0 = *(const bf16x8*)(brow0 + bh);
    bf16x8 cb1 = *(const bf16x8*)(brow0 + bh + 8);
    bf16x8 cb2 = *(const bf16x8*)(brow1 + bh);
    bf16x8 cb3 = *(const bf16x8*)(brow1 + bh + 8);

    f32x8 acc0 = {}, acc1 = {};
    const int iters = K >> 5;
    for (int it = 0; it < iters; ++it) {
        const int k0  = it << 5;
        const int cur = it & 1;
        const int has_next = (it + 1) < iters;

        if (has_next) {
            async_copy_A(X, m0, K, k0 + 32, Ash[cur ^ 1], tid);  // overlap with compute
            wait_async_le(1, tid);   // chunk 'it' done (in-order); next may fly
        } else {
            wait_async_le(0, tid);
        }
        __syncthreads();             // publish Ash[cur] to all 8 waves

        // issue B loads for next iteration BEFORE consuming current fragments
        bf16x8 nb0, nb1, nb2, nb3;
        if (has_next) {
            const int kn = k0 + 32 + bh;
            nb0 = *(const bf16x8*)(brow0 + kn);
            nb1 = *(const bf16x8*)(brow0 + kn + 8);
            nb2 = *(const bf16x8*)(brow1 + kn);
            nb3 = *(const bf16x8*)(brow1 + kn + 8);
        }

        // A fragment from LDS (ds_load_b128 x2)
        bf16x8 alo = *(const bf16x8*)&Ash[cur][r * 32 + ah];
        bf16x8 ahi = *(const bf16x8*)&Ash[cur][r * 32 + ah + 8];
        bf16x16 a, b0, b1;
#pragma unroll
        for (int i = 0; i < 8; ++i) {
            a[i]  = alo[i]; a[i + 8]  = ahi[i];
            b0[i] = cb0[i]; b0[i + 8] = cb1[i];
            b1[i] = cb2[i]; b1[i + 8] = cb3[i];
        }
        acc0 = __builtin_amdgcn_wmma_f32_16x16x32_bf16(false, a, false, b0,
                                                       (short)0, acc0, false, false);
        acc1 = __builtin_amdgcn_wmma_f32_16x16x32_bf16(false, a, false, b1,
                                                       (short)0, acc1, false, false);
        __syncthreads();             // all reads of Ash[cur] done before reuse

        if (has_next) { cb0 = nb0; cb1 = nb1; cb2 = nb2; cb3 = nb3; }
    }

    // C/D layout: VGPR v -> row m0 + v + 8*hsel, column tile_base + (lane&15)
    const int mb = m0 + hsel * 8;
    if (valid0) {
        const int n = n0 + r;
        const float bn = bias ? bias[n] : 0.0f;
#pragma unroll
        for (int v = 0; v < 8; ++v) {
            float y = acc0[v] + bn;
            if (relu) y = fmaxf(y, 0.0f);
            size_t off = (size_t)(mb + v) * Nc + n;
            if (Yf)  Yf[off]  = y;
            if (Ybf) Ybf[off] = f2bf(y);
        }
    }
    if (valid1) {
        const int n = n0 + 16 + r;
        const float bn = bias ? bias[n] : 0.0f;
#pragma unroll
        for (int v = 0; v < 8; ++v) {
            float y = acc1[v] + bn;
            if (relu) y = fmaxf(y, 0.0f);
            size_t off = (size_t)(mb + v) * Nc + n;
            if (Yf)  Yf[off]  = y;
            if (Ybf) Ybf[off] = f2bf(y);
        }
    }
}

// ---------------------------------------------------------------------------
// Per-row top-32 of sim[row, 0..N). One 256-thread block per row.
// Per-thread register top-32 then LDS merge (64KB LDS of 320KB WGP budget).
// ---------------------------------------------------------------------------
__global__ __launch_bounds__(256)
void k_topk32(const float* __restrict__ sim, int N,
              float* __restrict__ tv, int* __restrict__ ti) {
    __shared__ float sv[256 * KTOP];
    __shared__ int   si[256 * KTOP];
    __shared__ float rmax[256];
    __shared__ int   rarg[256];

    const int row = blockIdx.x, tid = threadIdx.x;
    const float* __restrict__ s = sim + (size_t)row * N;

    float lv[KTOP]; int li[KTOP];
#pragma unroll
    for (int k = 0; k < KTOP; ++k) { lv[k] = -3.4e38f; li[k] = 0; }
    float mn = -3.4e38f; int mp = 0;

    for (int n = tid; n < N; n += 256) {
        float v = s[n];
        if (v > mn) {
            lv[mp] = v; li[mp] = n;
            mn = lv[0]; mp = 0;
#pragma unroll
            for (int k = 1; k < KTOP; ++k)
                if (lv[k] < mn) { mn = lv[k]; mp = k; }
        }
    }
#pragma unroll
    for (int k = 0; k < KTOP; ++k) { sv[tid * KTOP + k] = lv[k]; si[tid * KTOP + k] = li[k]; }
    __syncthreads();

    for (int sel = 0; sel < KTOP; ++sel) {
        float bm = -3.4e38f; int ba = tid * KTOP;
#pragma unroll
        for (int k = 0; k < KTOP; ++k) {
            float v = sv[tid * KTOP + k];
            if (v > bm) { bm = v; ba = tid * KTOP + k; }
        }
        rmax[tid] = bm; rarg[tid] = ba;
        __syncthreads();
        for (int off = 128; off > 0; off >>= 1) {
            if (tid < off && rmax[tid + off] > rmax[tid]) {
                rmax[tid] = rmax[tid + off]; rarg[tid] = rarg[tid + off];
            }
            __syncthreads();
        }
        if (tid == 0) {
            tv[(size_t)row * KTOP + sel] = rmax[0];
            ti[(size_t)row * KTOP + sel] = si[rarg[0]];
            sv[rarg[0]] = -3.4e38f;
        }
        __syncthreads();
    }
}

// ---------------------------------------------------------------------------
// softmax(vals/SCALE) then weighted gather-sum over bf16 V bank rows
// ---------------------------------------------------------------------------
__global__ __launch_bounds__(256)
void k_attend(const float* __restrict__ tv, const int* __restrict__ ti,
              const unsigned short* __restrict__ Vbf,
              float* __restrict__ out, int D) {
    __shared__ float w[KTOP];
    __shared__ int   idx[KTOP];
    const int row = blockIdx.x, tid = threadIdx.x;
    if (tid == 0) {
        float mx = -3.4e38f;
        for (int k = 0; k < KTOP; ++k) mx = fmaxf(mx, tv[(size_t)row * KTOP + k]);
        float sum = 0.0f;
        for (int k = 0; k < KTOP; ++k) {
            float e = __expf((tv[(size_t)row * KTOP + k] - mx) * INV_SCALE);
            w[k] = e; sum += e;
        }
        float inv = 1.0f / sum;
        for (int k = 0; k < KTOP; ++k) { w[k] *= inv; idx[k] = ti[(size_t)row * KTOP + k]; }
    }
    __syncthreads();
    for (int d = tid; d < D; d += 256) {
        float acc = 0.0f;
#pragma unroll
        for (int k = 0; k < KTOP; ++k)
            acc += w[k] * bf2f(Vbf[(size_t)idx[k] * D + d]);
        out[(size_t)row * D + d] = acc;
    }
}

// ---------------------------------------------------------------------------
// cosine head: (dot(l2n(p), l2n(m)) + 1) * 0.5 per row
// ---------------------------------------------------------------------------
__global__ __launch_bounds__(256)
void k_cos_head(const float* __restrict__ p2, const float* __restrict__ m2,
                float* __restrict__ out, int D) {
    __shared__ float sdot[256], sn1[256], sn2[256];
    const int row = blockIdx.x, tid = threadIdx.x;
    float dot = 0.f, n1 = 0.f, n2 = 0.f;
    for (int d = tid; d < D; d += 256) {
        float a = p2[(size_t)row * D + d], b = m2[(size_t)row * D + d];
        dot += a * b; n1 += a * a; n2 += b * b;
    }
    sdot[tid] = dot; sn1[tid] = n1; sn2[tid] = n2;
    __syncthreads();
    for (int off = 128; off > 0; off >>= 1) {
        if (tid < off) {
            sdot[tid] += sdot[tid + off];
            sn1[tid]  += sn1[tid + off];
            sn2[tid]  += sn2[tid + off];
        }
        __syncthreads();
    }
    if (tid == 0) {
        float na = fmaxf(sqrtf(sn1[0]), 1e-12f);
        float nb = fmaxf(sqrtf(sn2[0]), 1e-12f);
        out[row] = (sdot[0] / (na * nb) + 1.0f) * 0.5f;
    }
}

// ---------------------------------------------------------------------------
// Orchestration
// ---------------------------------------------------------------------------
extern "C" void kernel_launch(void* const* d_in, const int* in_sizes, int n_in,
                              void* d_out, int out_size, void* d_ws, size_t ws_size,
                              hipStream_t stream) {
    (void)in_sizes; (void)n_in; (void)out_size; (void)ws_size;

    const float* prot_emb = (const float*)d_in[0];
    const float* comp_emb = (const float*)d_in[1];
    const float* prot_bank = (const float*)d_in[2];
    const float* comp_bank = (const float*)d_in[3];
    const float* pq_W = (const float*)d_in[4];  const float* pq_b = (const float*)d_in[5];
    const float* mq_W = (const float*)d_in[6];  const float* mq_b = (const float*)d_in[7];
    const float* pk_W = (const float*)d_in[8];  const float* pk_b = (const float*)d_in[9];
    const float* mk_W = (const float*)d_in[10]; const float* mk_b = (const float*)d_in[11];
    const float* pv_W = (const float*)d_in[12]; const float* pv_b = (const float*)d_in[13];
    const float* mv_W = (const float*)d_in[14]; const float* mv_b = (const float*)d_in[15];
    const float* po_W1 = (const float*)d_in[16]; const float* po_b1 = (const float*)d_in[17];
    const float* po_W2 = (const float*)d_in[18]; const float* po_b2 = (const float*)d_in[19];
    const float* mo_W1 = (const float*)d_in[20]; const float* mo_b1 = (const float*)d_in[21];
    const float* mo_W2 = (const float*)d_in[22]; const float* mo_b2 = (const float*)d_in[23];

    // ---- workspace linear allocator ----
    char* wsp = (char*)d_ws;
    size_t off = 0;
    auto alloc = [&](size_t bytes) -> void* {
        void* p = wsp + off;
        off += (bytes + 255) & ~(size_t)255;
        return p;
    };

    unsigned short* bankP_bf = (unsigned short*)alloc((size_t)NN * DD * 2);
    unsigned short* bankM_bf = (unsigned short*)alloc((size_t)NN * DD * 2);
    unsigned short* pkb_bf   = (unsigned short*)alloc((size_t)NN * DD * 2);
    unsigned short* mkb_bf   = (unsigned short*)alloc((size_t)NN * DD * 2);
    unsigned short* pvb_bf   = (unsigned short*)alloc((size_t)NN * DD * 2);
    unsigned short* mvb_bf   = (unsigned short*)alloc((size_t)NN * DD * 2);
    unsigned short* wpq = (unsigned short*)alloc((size_t)DD * DD * 2);
    unsigned short* wmq = (unsigned short*)alloc((size_t)DD * DD * 2);
    unsigned short* wpk = (unsigned short*)alloc((size_t)DD * DD * 2);
    unsigned short* wmk = (unsigned short*)alloc((size_t)DD * DD * 2);
    unsigned short* wpv = (unsigned short*)alloc((size_t)DD * DD * 2);
    unsigned short* wmv = (unsigned short*)alloc((size_t)DD * DD * 2);
    unsigned short* wpo1 = (unsigned short*)alloc((size_t)DD * 2 * DD * 2);
    unsigned short* wmo1 = (unsigned short*)alloc((size_t)DD * 2 * DD * 2);
    unsigned short* wpo2 = (unsigned short*)alloc((size_t)DD * DD * 2);
    unsigned short* wmo2 = (unsigned short*)alloc((size_t)DD * DD * 2);
    unsigned short* embP_bf = (unsigned short*)alloc((size_t)BB * DD * 2);
    unsigned short* embM_bf = (unsigned short*)alloc((size_t)BB * DD * 2);
    unsigned short* pq_bf = (unsigned short*)alloc((size_t)BB * DD * 2);
    unsigned short* mq_bf = (unsigned short*)alloc((size_t)BB * DD * 2);
    float* simbuf = (float*)alloc((size_t)BB * NN * 4);
    float* tvf    = (float*)alloc((size_t)BB * KTOP * 4);
    int*   tii    = (int*)alloc((size_t)BB * KTOP * 4);
    float* sim_mp = (float*)alloc((size_t)BB * DD * 4);
    float* sim_pm = (float*)alloc((size_t)BB * DD * 4);
    unsigned short* cat_p = (unsigned short*)alloc((size_t)BB * 2 * DD * 2);
    unsigned short* cat_m = (unsigned short*)alloc((size_t)BB * 2 * DD * 2);
    unsigned short* hid_p = (unsigned short*)alloc((size_t)BB * DD * 2);
    unsigned short* hid_m = (unsigned short*)alloc((size_t)BB * DD * 2);
    float* p2f = (float*)alloc((size_t)BB * DD * 4);
    float* m2f = (float*)alloc((size_t)BB * DD * 4);

    auto cvt = [&](const float* src, unsigned short* dst, size_t n) {
        k_f32_to_bf16<<<(unsigned)((n + 255) / 256), 256, 0, stream>>>(src, dst, n);
    };

    // 1) convert weights / banks / embeddings to bf16
    cvt(pq_W, wpq, (size_t)DD * DD);  cvt(mq_W, wmq, (size_t)DD * DD);
    cvt(pk_W, wpk, (size_t)DD * DD);  cvt(mk_W, wmk, (size_t)DD * DD);
    cvt(pv_W, wpv, (size_t)DD * DD);  cvt(mv_W, wmv, (size_t)DD * DD);
    cvt(po_W1, wpo1, (size_t)DD * 2 * DD); cvt(mo_W1, wmo1, (size_t)DD * 2 * DD);
    cvt(po_W2, wpo2, (size_t)DD * DD);     cvt(mo_W2, wmo2, (size_t)DD * DD);
    cvt(prot_bank, bankP_bf, (size_t)NN * DD);
    cvt(comp_bank, bankM_bf, (size_t)NN * DD);
    cvt(prot_emb, embP_bf, (size_t)BB * DD);
    cvt(comp_emb, embM_bf, (size_t)BB * DD);

    auto gemm = [&](const unsigned short* X, const unsigned short* W, const float* bias,
                    float* Yf, unsigned short* Ybf, int M, int Nc, int K, int relu) {
        dim3 grid((unsigned)((Nc + 255) / 256), (unsigned)(M / 16));
        k_wmma_gemm_bf16<<<grid, 256, 0, stream>>>(
            (const bf16_t*)X, (const bf16_t*)W, bias, Yf, Ybf, M, Nc, K, relu);
    };

    // 2) query projections  q = emb @ Wq^T + b  -> bf16
    gemm(embP_bf, wpq, pq_b, nullptr, pq_bf, BB, DD, DD, 0);
    gemm(embM_bf, wmq, mq_b, nullptr, mq_bf, BB, DD, DD, 0);

    // 3) bank K/V projections -> bf16 banks
    gemm(bankP_bf, wpk, pk_b, nullptr, pkb_bf, NN, DD, DD, 0);
    gemm(bankM_bf, wmk, mk_b, nullptr, mkb_bf, NN, DD, DD, 0);
    gemm(bankP_bf, wpv, pv_b, nullptr, pvb_bf, NN, DD, DD, 0);
    gemm(bankM_bf, wmv, mv_b, nullptr, mvb_bf, NN, DD, DD, 0);

    // 4) direction mp: compound queries over protein bank
    gemm(mq_bf, pkb_bf, nullptr, simbuf, nullptr, BB, NN, DD, 0);
    k_topk32<<<BB, 256, 0, stream>>>(simbuf, NN, tvf, tii);
    k_attend<<<BB, 256, 0, stream>>>(tvf, tii, pvb_bf, sim_mp, DD);

    // 5) direction pm: protein queries over compound bank (reuse simbuf)
    gemm(pq_bf, mkb_bf, nullptr, simbuf, nullptr, BB, NN, DD, 0);
    k_topk32<<<BB, 256, 0, stream>>>(simbuf, NN, tvf, tii);
    k_attend<<<BB, 256, 0, stream>>>(tvf, tii, mvb_bf, sim_pm, DD);

    // 6) output MLPs
    k_concat_bf16<<<BB, 256, 0, stream>>>(prot_emb, sim_pm, cat_p, DD);
    k_concat_bf16<<<BB, 256, 0, stream>>>(comp_emb, sim_mp, cat_m, DD);
    gemm(cat_p, wpo1, po_b1, nullptr, hid_p, BB, DD, 2 * DD, 1);
    gemm(cat_m, wmo1, mo_b1, nullptr, hid_m, BB, DD, 2 * DD, 1);
    gemm(hid_p, wpo2, po_b2, p2f, nullptr, BB, DD, DD, 0);
    gemm(hid_m, wmo2, mo_b2, m2f, nullptr, BB, DD, DD, 0);

    // 7) cosine similarity head
    k_cos_head<<<BB, 256, 0, stream>>>(p2f, m2f, (float*)d_out, DD);
}
